// GATConv_80942953660640
// MI455X (gfx1250) — matrix operand order repaired and checked by reading, not compiled, not run on hardware
//
#include <hip/hip_runtime.h>

typedef __attribute__((ext_vector_type(16))) __bf16 v16bf;
typedef __attribute__((ext_vector_type(8)))  float  v8f;
typedef __attribute__((ext_vector_type(4)))  float  v4f;

#define NN        8192
#define IN_CH     128
#define OUT_CH    64
#define HEADS     2
#define NEG_SLOPE 0.2f

// ---------------------------------------------------------------------------
// Phase 1: Xp = X @ W  -> bf16, stored k-major per (head, j) so phase-2 WMMA-B
// fragments are contiguous 32B loads. Also X0/X1 attention halves in fp32.
// One block per node i, 128 threads = (j, head) pairs.
// ---------------------------------------------------------------------------
__global__ __launch_bounds__(128) void gat_project(
    const float* __restrict__ X, const float* __restrict__ W,
    const float* __restrict__ attn,
    __bf16* __restrict__ Xpb, float* __restrict__ X0t, float* __restrict__ X1t)
{
    __shared__ float Xs[IN_CH];
    __shared__ float red0[HEADS];
    __shared__ float red1[HEADS];
    const int i = blockIdx.x;
    const int t = threadIdx.x;
    Xs[t] = X[(size_t)i * IN_CH + t];
    if (t < HEADS) { red0[t] = 0.f; red1[t] = 0.f; }
    __syncthreads();

    const int j = t >> 1;        // output channel 0..63
    const int l = t & 1;         // head
    float acc = 0.f;
#pragma unroll 8
    for (int k = 0; k < IN_CH; ++k)
        acc += Xs[k] * W[(k * OUT_CH + j) * HEADS + l];   // coalesced over t per k

    // bf16 projected feature, k-major: Xpb[(l*64 + j)*N + i]
    Xpb[(size_t)(l * OUT_CH + j) * NN + i] = (__bf16)acc;

    // attention halves: X0[i,l] = sum_j Xp*attn[j,l] ; X1 uses attn[64+j,l]
    atomicAdd(&red0[l], acc * attn[j * HEADS + l]);
    atomicAdd(&red1[l], acc * attn[(OUT_CH + j) * HEADS + l]);
    __syncthreads();
    if (t < HEADS) {
        X0t[(size_t)t * NN + i] = red0[t];
        X1t[(size_t)t * NN + i] = red1[t];
    }
}

// ---------------------------------------------------------------------------
// Phase 2: single streaming pass over A. For each 16-row tile (block) each of
// the 4 waves covers a 2048-wide K chunk: build the bf16 Aw WMMA-A fragment
// (fused exp(leaky_relu)) and accumulate C[16x64] per head with
// v_wmma_f32_16x16x32_bf16 + exact fp32 row sums. LDS reduction + finalize.
// ---------------------------------------------------------------------------
__global__ __launch_bounds__(128) void gat_aggregate(
    const float* __restrict__ A, const __bf16* __restrict__ Xpb,
    const float* __restrict__ X0t, const float* __restrict__ X1t,
    const float* __restrict__ bias, float* __restrict__ out)
{
    __shared__ float Cred[HEADS][16][OUT_CH];   // 8 KB
    __shared__ float RSred[HEADS][16];

    const int tid  = threadIdx.x;
    const int lane = tid & 31;
    const int wave = tid >> 5;
    const int half = lane >> 4;      // K-half selector per ISA A/B layouts
    const int m    = lane & 15;      // row within tile (A frag) / col n (B frag)
    const int i0   = blockIdx.x * 16;

    for (int idx = tid; idx < HEADS * 16 * OUT_CH; idx += 128)
        ((float*)Cred)[idx] = 0.f;
    if (tid < HEADS * 16) ((float*)RSred)[tid] = 0.f;
    __syncthreads();

    const float x0h0 = X0t[0 * NN + i0 + m];
    const float x0h1 = X0t[1 * NN + i0 + m];

    v8f zero = {0.f,0.f,0.f,0.f,0.f,0.f,0.f,0.f};
    v8f acc0[4], acc1[4];
#pragma unroll
    for (int tc = 0; tc < 4; ++tc) { acc0[tc] = zero; acc1[tc] = zero; }
    float rs0 = 0.f, rs1 = 0.f;

    const int   kbeg = wave * (NN / 4);
    const int   kend = kbeg + (NN / 4);
    const float* __restrict__ Arow = A + (size_t)(i0 + m) * NN;
    const float* __restrict__ X1h0 = X1t;
    const float* __restrict__ X1h1 = X1t + NN;

    for (int k0 = kbeg; k0 < kend; k0 += 32) {
        // --- A-fragment source values: 16-bit A 16x32 k order used here (not the ISA table order):
        // lane(half,m): elements 0..15 -> K = 16*half + e (contiguous, matches B)
        const int o1 = k0 + half * 16;       // run 1 (float4-aligned)
        const int o2 = o1 + 8;               // run 2
        v4f va0 = __builtin_nontemporal_load((const v4f*)(Arow + o1));
        v4f va1 = __builtin_nontemporal_load((const v4f*)(Arow + o1 + 4));
        v4f va2 = __builtin_nontemporal_load((const v4f*)(Arow + o2));
        v4f va3 = __builtin_nontemporal_load((const v4f*)(Arow + o2 + 4));
        v4f xq0 = *(const v4f*)(X1h0 + o1);
        v4f xq1 = *(const v4f*)(X1h0 + o1 + 4);
        v4f xq2 = *(const v4f*)(X1h0 + o2);
        v4f xq3 = *(const v4f*)(X1h0 + o2 + 4);
        v4f xp0 = *(const v4f*)(X1h1 + o1);
        v4f xp1 = *(const v4f*)(X1h1 + o1 + 4);
        v4f xp2 = *(const v4f*)(X1h1 + o2);
        v4f xp3 = *(const v4f*)(X1h1 + o2 + 4);

        float av[16], x1a[16], x1b[16];
        av[0]=va0.x; av[1]=va0.y; av[2]=va0.z; av[3]=va0.w;
        av[4]=va1.x; av[5]=va1.y; av[6]=va1.z; av[7]=va1.w;
        av[8]=va2.x; av[9]=va2.y; av[10]=va2.z; av[11]=va2.w;
        av[12]=va3.x; av[13]=va3.y; av[14]=va3.z; av[15]=va3.w;
        x1a[0]=xq0.x; x1a[1]=xq0.y; x1a[2]=xq0.z; x1a[3]=xq0.w;
        x1a[4]=xq1.x; x1a[5]=xq1.y; x1a[6]=xq1.z; x1a[7]=xq1.w;
        x1a[8]=xq2.x; x1a[9]=xq2.y; x1a[10]=xq2.z; x1a[11]=xq2.w;
        x1a[12]=xq3.x; x1a[13]=xq3.y; x1a[14]=xq3.z; x1a[15]=xq3.w;
        x1b[0]=xp0.x; x1b[1]=xp0.y; x1b[2]=xp0.z; x1b[3]=xp0.w;
        x1b[4]=xp1.x; x1b[5]=xp1.y; x1b[6]=xp1.z; x1b[7]=xp1.w;
        x1b[8]=xp2.x; x1b[9]=xp2.y; x1b[10]=xp2.z; x1b[11]=xp2.w;
        x1b[12]=xp3.x; x1b[13]=xp3.y; x1b[14]=xp3.z; x1b[15]=xp3.w;

        // --- fused edge weight: aw = A * exp(leaky_relu(x0 + x1)) per head
        v16bf af0, af1;
#pragma unroll
        for (int e = 0; e < 16; ++e) {
            float a = av[e];
            float s0 = x0h0 + x1a[e];
            float t0 = fmaxf(s0, 0.f) + NEG_SLOPE * fminf(s0, 0.f);
            float w0 = a * __expf(t0);
            rs0 += w0;                 // exact fp32 row-sum (lane pair m,m+16 covers all K)
            af0[e] = (__bf16)w0;
            float s1 = x0h1 + x1b[e];
            float t1 = fmaxf(s1, 0.f) + NEG_SLOPE * fminf(s1, 0.f);
            float w1 = a * __expf(t1);
            rs1 += w1;
            af1[e] = (__bf16)w1;
        }

        // --- B fragments: Xpb[(l*64 + j0+n)][k], lane loads K = 16*half + e (contiguous)
        const size_t kb = (size_t)k0 + half * 16;
#pragma unroll
        for (int tc = 0; tc < 4; ++tc) {
            v16bf b0 = *(const v16bf*)(Xpb + ((size_t)(0 * OUT_CH + tc * 16 + m)) * NN + kb);
            acc0[tc] = __builtin_amdgcn_wmma_f32_16x16x32_bf16(
                false, af0, false, b0, (short)0, acc0[tc], false, false);
            v16bf b1 = *(const v16bf*)(Xpb + ((size_t)(1 * OUT_CH + tc * 16 + m)) * NN + kb);
            acc1[tc] = __builtin_amdgcn_wmma_f32_16x16x32_bf16(
                false, af1, false, b1, (short)0, acc1[tc], false, false);
        }
    }

    // combine lane halves of the fp32 row sums (lanes m and m+16 share row m)
    rs0 += __shfl_xor(rs0, 16, 32);
    rs1 += __shfl_xor(rs1, 16, 32);

    // --- cross-wave reduction: C/D layout -> row = r + 8*half, col = tc*16 + n
#pragma unroll
    for (int tc = 0; tc < 4; ++tc)
#pragma unroll
        for (int r = 0; r < 8; ++r) {
            int mr = r + half * 8;
            atomicAdd(&Cred[0][mr][tc * 16 + m], acc0[tc][r]);
            atomicAdd(&Cred[1][mr][tc * 16 + m], acc1[tc][r]);
        }
    if (half == 0) {
        atomicAdd(&RSred[0][m], rs0);
        atomicAdd(&RSred[1][m], rs1);
    }
    __syncthreads();

    // --- finalize: out[i, l, j] = C * 1/(rowsum + 1e-3) + bias[l, j]
    for (int idx = tid; idx < 16 * HEADS * OUT_CH; idx += 128) {
        int mm  = idx >> 7;
        int rem = idx & 127;
        int l   = rem >> 6;
        int j   = rem & 63;
        float dinv = 1.f / (RSred[l][mm] + 0.001f);
        out[(size_t)(i0 + mm) * (HEADS * OUT_CH) + l * OUT_CH + j] =
            Cred[l][mm][j] * dinv + bias[l * OUT_CH + j];
    }
}

// ---------------------------------------------------------------------------
extern "C" void kernel_launch(void* const* d_in, const int* in_sizes, int n_in,
                              void* d_out, int out_size, void* d_ws, size_t ws_size,
                              hipStream_t stream)
{
    const float* A    = (const float*)d_in[0];
    const float* X    = (const float*)d_in[1];
    const float* W    = (const float*)d_in[2];
    const float* attn = (const float*)d_in[3];
    const float* b    = (const float*)d_in[4];
    float* out = (float*)d_out;

    char* ws = (char*)d_ws;
    __bf16* Xpb = (__bf16*)ws;                                  // 2 MB
    float*  X0t = (float*)(ws + (size_t)HEADS * OUT_CH * NN * sizeof(__bf16));
    float*  X1t = X0t + (size_t)HEADS * NN;                     // 64 KB each

    gat_project<<<NN, IN_CH, 0, stream>>>(X, W, attn, Xpb, X0t, X1t);
    gat_aggregate<<<NN / 16, 128, 0, stream>>>(A, Xpb, X0t, X1t, b, out);
}
